// RelationTreeLSTM_31164282700373
// MI455X (gfx1250) — compile-verified, matrix operand order checked by reading, not run on hardware
//
#include <hip/hip_runtime.h>
#include <hip/hip_bf16.h>

// ---------------- problem constants (from reference source) ----------------
#define N_NODES 4681          // (8^5-1)/7
#define NP1     4682          // + sentinel row
#define NPAD    4688          // 293*16
#define HDIM    300
#define HP      304           // H padded to multiple of 16
#define KP      320           // K padded to 10 chunks of 32 (f16 WMMA K)
#define C3      900
#define C3P     912           // 3H padded to multiple of 16
#define LVLW    4096
#define TOUT    42

typedef __attribute__((ext_vector_type(16))) _Float16 v16h;
typedef __attribute__((ext_vector_type(8)))  _Float16 v8h;
typedef __attribute__((ext_vector_type(8)))  float    v8f;

union HV16 { v16h v; v8h h[2]; };

// A/B fragment loader for V_WMMA_F32_16X16X32_F16.
// Matrix stored row-major with row stride KP halves; fragment rows r0..r0+15.
// ISA 16-bit A layout: lanes 0-15 row=lane, K = {kc*32+0..7, kc*32+16..23};
// lanes 16-31 same rows? no: row = lane-16, K = {kc*32+8..15, kc*32+24..31}.
__device__ __forceinline__ v16h load_frag(const _Float16* base, int r0, int kc, int lane) {
  const _Float16* rp = base + (size_t)(r0 + (lane & 15)) * KP + kc * 32 + (lane >> 4) * 8;
  HV16 u;
  u.h[0] = *(const v8h*)(rp);        // 16B load
  u.h[1] = *(const v8h*)(rp + 16);   // 16B load
  return u.v;
}

__device__ __forceinline__ v8f wmma_f16(v16h a, v16h b, v8f c) {
  // D = A(16x32 f16) x B(32x16 f16) + C(16x16 f32)
  return __builtin_amdgcn_wmma_f32_16x16x32_f16(false, a, false, b, (short)0, c, false, false);
}

__device__ __forceinline__ float sigf(float x) { return 1.0f / (1.0f + __expf(-x)); }

// ---------------- prep kernels ----------------
__global__ void k_zero(float* __restrict__ p, int n) {
  int i = blockIdx.x * blockDim.x + threadIdx.x;
  if (i < n) p[i] = 0.0f;
}

// src: [300][ncols] row-major f32  ->  dst: [npad][KP] f16, dst[n][k] = src[k][n]
__global__ void k_cvt_wt(const float* __restrict__ src, _Float16* __restrict__ dst,
                         int ncols, int npad) {
  int i = blockIdx.x * blockDim.x + threadIdx.x;
  int tot = npad * KP;
  if (i >= tot) return;
  int n = i / KP, k = i % KP;
  float v = (n < ncols && k < HDIM) ? src[(size_t)k * ncols + n] : 0.0f;
  dst[i] = (_Float16)v;
}

// x_h[m][k] = embed[tokens[m]][k] (f16), zero-padded rows/cols
__global__ void k_gather_x(const int* __restrict__ tokens, const float* __restrict__ embed,
                           _Float16* __restrict__ xh) {
  int i = blockIdx.x * blockDim.x + threadIdx.x;
  const int tot = NPAD * KP;
  if (i >= tot) return;
  int m = i / KP, k = i % KP;
  float v = (m < N_NODES && k < HDIM) ? embed[(size_t)tokens[m] * HDIM + k] : 0.0f;
  xh[i] = (_Float16)v;
}

// ---------------- batched input GEMM: xiou = x@W_ioux + b, fxb = x@W_fx + b ----
__global__ void __launch_bounds__(256)
k_xgemm(const _Float16* __restrict__ xh,
        const _Float16* __restrict__ wt_ioux, const _Float16* __restrict__ wt_fx,
        const float* __restrict__ b_ioux, const float* __restrict__ b_fx,
        float* __restrict__ xiou, float* __restrict__ fxb) {
  const int m0   = blockIdx.x * 16;
  const int lane = threadIdx.x & 31;
  const int wave = threadIdx.x >> 5;
  const int nl   = lane & 15;
  const int ms   = (lane >> 4) * 8;

  for (int t = wave; t < 57 + 19; t += 8) {   // 57 iou tiles + 19 fx tiles
    v8f acc = {};
    if (t < 57) {
      const int n0 = t * 16;
#pragma unroll
      for (int kc = 0; kc < 10; ++kc) {
        v16h a = load_frag(xh, m0, kc, lane);
        v16h b = load_frag(wt_ioux, n0, kc, lane);
        acc = wmma_f16(a, b, acc);
      }
      const int n = n0 + nl;
#pragma unroll
      for (int rr = 0; rr < 8; ++rr) {
        const int m = m0 + ms + rr;
        if (m < N_NODES && n < C3) xiou[(size_t)m * C3P + n] = acc[rr] + b_ioux[n];
      }
    } else {
      const int n0 = (t - 57) * 16;
#pragma unroll
      for (int kc = 0; kc < 10; ++kc) {
        v16h a = load_frag(xh, m0, kc, lane);
        v16h b = load_frag(wt_fx, n0, kc, lane);
        acc = wmma_f16(a, b, acc);
      }
      const int n = n0 + nl;
#pragma unroll
      for (int rr = 0; rr < 8; ++rr) {
        const int m = m0 + ms + rr;
        if (m < N_NODES && n < HDIM) fxb[(size_t)m * HP + n] = acc[rr] + b_fx[n];
      }
    }
  }
}

// ---------------- fused per-level kernel (block = 16 level entries) ----------
__global__ void __launch_bounds__(256)
k_level(const int* __restrict__ lvl_row, const int* __restrict__ parent,
        const float* __restrict__ xiou, const float* __restrict__ fxb,
        const float* __restrict__ b_iouh, const float* __restrict__ b_fh,
        const _Float16* __restrict__ wt_iouh, const _Float16* __restrict__ wt_fh,
        float* __restrict__ h_sum, float* __restrict__ fc_sum,
        float* __restrict__ h_all, float* __restrict__ c_all) {
  __shared__ int s_idx[16];
  __shared__ int s_par[16];
  __shared__ __align__(16) _Float16 A_lds[16][KP];   // h_sum rows (f16)
  __shared__ __align__(16) _Float16 A2_lds[16][KP];  // h_new rows (f16)
  __shared__ float cn_lds[16][HP];
  __shared__ float hn_lds[16][HP];

  const int tid  = threadIdx.x;
  const int lane = tid & 31;
  const int wave = tid >> 5;
  const int nl   = lane & 15;
  const int ms   = (lane >> 4) * 8;

  if (tid < 16) {
    int node = lvl_row[blockIdx.x * 16 + tid];
    if (node < 0 || node > N_NODES) node = N_NODES;  // sentinel
    s_idx[tid] = node;
    s_par[tid] = (node < N_NODES) ? parent[node] : N_NODES;
  }
  __syncthreads();

  // stage 1: stage h_sum rows (f32 -> f16), zero A2 (incl. K pad cols)
  for (int i = tid; i < 16 * KP; i += 256) {
    int m = i / KP, k = i % KP;
    float hv = (k < HDIM) ? h_sum[(size_t)s_idx[m] * HP + k] : 0.0f;
    A_lds[m][k]  = (_Float16)hv;
    A2_lds[m][k] = (_Float16)0.0f;
  }
  __syncthreads();

  // stage 2+3: iou GEMM (i/o/u tiles together) + gate math on fragments
  for (int t = wave; t < 19; t += 8) {
    const int n0 = t * 16;
    v8f ci = {}, co = {}, cu = {};
#pragma unroll
    for (int kc = 0; kc < 10; ++kc) {
      v16h a  = load_frag(&A_lds[0][0], 0, kc, lane);
      v16h bi = load_frag(wt_iouh, n0,       kc, lane);
      v16h bo = load_frag(wt_iouh, 300 + n0, kc, lane);
      v16h bu = load_frag(wt_iouh, 600 + n0, kc, lane);
      ci = wmma_f16(a, bi, ci);
      co = wmma_f16(a, bo, co);
      cu = wmma_f16(a, bu, cu);
    }
    const int n = n0 + nl;
#pragma unroll
    for (int rr = 0; rr < 8; ++rr) {
      const int m = ms + rr;
      const int node = s_idx[m];
      if (n < HDIM && node < N_NODES) {
        const size_t xb = (size_t)node * C3P;
        float iv = ci[rr] + xiou[xb + n]       + b_iouh[n];
        float ov = co[rr] + xiou[xb + 300 + n] + b_iouh[300 + n];
        float uv = cu[rr] + xiou[xb + 600 + n] + b_iouh[600 + n];
        float cc = sigf(iv) * tanhf(uv) + fc_sum[(size_t)node * HP + n];
        float hh = sigf(ov) * tanhf(cc);
        cn_lds[m][n] = cc;
        hn_lds[m][n] = hh;
        A2_lds[m][n] = (_Float16)hh;
        c_all[(size_t)node * HP + n] = cc;
        h_all[(size_t)node * HP + n] = hh;
      }
    }
  }
  __syncthreads();

  // stage 4+5: f = sigmoid(h_new@W_fh + b_fh + fx[parent]); scatter-add to parent
  for (int t = wave; t < 19; t += 8) {
    const int n0 = t * 16;
    v8f cf = {};
#pragma unroll
    for (int kc = 0; kc < 10; ++kc) {
      v16h a = load_frag(&A2_lds[0][0], 0, kc, lane);
      v16h b = load_frag(wt_fh, n0, kc, lane);
      cf = wmma_f16(a, b, cf);
    }
    const int n = n0 + nl;
#pragma unroll
    for (int rr = 0; rr < 8; ++rr) {
      const int m = ms + rr;
      const int node = s_idx[m];
      if (n < HDIM && node < N_NODES) {
        const int p = s_par[m];
        float f = sigf(cf[rr] + b_fh[n] + fxb[(size_t)p * HP + n]);
        atomicAdd(&h_sum[(size_t)p * HP + n], hn_lds[m][n]);
        atomicAdd(&fc_sum[(size_t)p * HP + n], f * cn_lds[m][n]);
      }
    }
  }
}

// ---------------- output head: out = root_c@W_lin + b_lin ; root_h -----------
__global__ void k_out(const float* __restrict__ c_all, const float* __restrict__ h_all,
                      const float* __restrict__ W_lin, const float* __restrict__ b_lin,
                      float* __restrict__ out) {
  int t = threadIdx.x;
  if (t < TOUT) {
    float s = b_lin[t];
    for (int k = 0; k < HDIM; ++k) s += c_all[k] * W_lin[(size_t)k * TOUT + t];
    out[t] = s;
  } else if (t < TOUT + HDIM) {
    out[t] = h_all[t - TOUT];
  }
}

// ---------------- host launcher ----------------
extern "C" void kernel_launch(void* const* d_in, const int* in_sizes, int n_in,
                              void* d_out, int out_size, void* d_ws, size_t ws_size,
                              hipStream_t stream) {
  const int*   tokens = (const int*)  d_in[0];
  const int*   parent = (const int*)  d_in[1];
  const int*   lvl    = (const int*)  d_in[2];
  const float* embed  = (const float*)d_in[3];
  const float* W_ioux = (const float*)d_in[4];
  const float* b_ioux = (const float*)d_in[5];
  const float* W_iouh = (const float*)d_in[6];
  const float* b_iouh = (const float*)d_in[7];
  const float* W_fx   = (const float*)d_in[8];
  const float* b_fx   = (const float*)d_in[9];
  const float* W_fh   = (const float*)d_in[10];
  const float* b_fh   = (const float*)d_in[11];
  const float* W_lin  = (const float*)d_in[12];
  const float* b_lin  = (const float*)d_in[13];
  float* out = (float*)d_out;

  char* ws = (char*)d_ws;
  size_t off = 0;
  auto carve = [&](size_t bytes) -> char* {
    char* p = ws + off;
    off += (bytes + 255) & ~(size_t)255;
    return p;
  };

  _Float16* wt_ioux = (_Float16*)carve((size_t)C3P * KP * 2);
  _Float16* wt_iouh = (_Float16*)carve((size_t)C3P * KP * 2);
  _Float16* wt_fx   = (_Float16*)carve((size_t)HP  * KP * 2);
  _Float16* wt_fh   = (_Float16*)carve((size_t)HP  * KP * 2);
  _Float16* xh      = (_Float16*)carve((size_t)NPAD * KP * 2);
  float* xiou   = (float*)carve((size_t)NP1 * C3P * 4);
  float* fxb    = (float*)carve((size_t)NP1 * HP * 4);
  float* h_sum  = (float*)carve((size_t)NP1 * HP * 4);
  float* fc_sum = (float*)carve((size_t)NP1 * HP * 4);
  float* h_all  = (float*)carve((size_t)NP1 * HP * 4);
  float* c_all  = (float*)carve((size_t)NP1 * HP * 4);

  // re-initialize state every call (graph-replay safe)
  const int zn = NP1 * HP;
  k_zero<<<(zn + 255) / 256, 256, 0, stream>>>(h_sum, zn);
  k_zero<<<(zn + 255) / 256, 256, 0, stream>>>(fc_sum, zn);
  k_zero<<<(C3P + 255) / 256, 256, 0, stream>>>(xiou + (size_t)N_NODES * C3P, C3P); // zero sentinel row
  k_zero<<<(HP + 255) / 256, 256, 0, stream>>>(fxb + (size_t)N_NODES * HP, HP);     // zero sentinel row

  const int wtTot1 = C3P * KP, wtTot2 = HP * KP;
  k_cvt_wt<<<(wtTot1 + 255) / 256, 256, 0, stream>>>(W_ioux, wt_ioux, C3, C3P);
  k_cvt_wt<<<(wtTot1 + 255) / 256, 256, 0, stream>>>(W_iouh, wt_iouh, C3, C3P);
  k_cvt_wt<<<(wtTot2 + 255) / 256, 256, 0, stream>>>(W_fx, wt_fx, HDIM, HP);
  k_cvt_wt<<<(wtTot2 + 255) / 256, 256, 0, stream>>>(W_fh, wt_fh, HDIM, HP);

  const int gxTot = NPAD * KP;
  k_gather_x<<<(gxTot + 255) / 256, 256, 0, stream>>>(tokens, embed, xh);

  k_xgemm<<<NPAD / 16, 256, 0, stream>>>(xh, wt_ioux, wt_fx, b_ioux, b_fx, xiou, fxb);

  // levels deepest -> root; sizes 4096, 512, 64, 8, 1 (rows padded with N)
  const int lvl_blocks[5] = {256, 32, 4, 1, 1};
  for (int d = 0; d < 5; ++d) {
    k_level<<<lvl_blocks[d], 256, 0, stream>>>(lvl + d * LVLW, parent,
                                               xiou, fxb, b_iouh, b_fh,
                                               wt_iouh, wt_fh,
                                               h_sum, fc_sum, h_all, c_all);
  }

  k_out<<<1, 384, 0, stream>>>(c_all, h_all, W_lin, b_lin, out);
}
